// Controller_41386304864570
// MI455X (gfx1250) — matrix operand order repaired and checked by reading, not compiled, not run on hardware
//
#include <hip/hip_runtime.h>
#include <hip/hip_bf16.h>
#include <math.h>

#define PN 4
#define LN 32
#define HN 1024
#define NSLOT (PN * (LN + 1)) /* 132 anchor slots */

typedef __attribute__((ext_vector_type(16))) _Float16 v16h;
typedef __attribute__((ext_vector_type(8)))  float    v8f;

// ---------------------------------------------------------------- helpers ---

__device__ __forceinline__ v8f wmma_f16(v16h a, v16h b, v8f c) {
  // D = A(16x32 f16) * B(32x16 f16) + C(16x16 f32)
  return __builtin_amdgcn_wmma_f32_16x16x32_f16(
      /*neg_a=*/false, a, /*neg_b=*/false, b,
      /*c_mod=*/(short)0, c, /*reuse_a=*/false, /*reuse_b=*/false);
}

// A-matrix fragment for this lane: row pointer, K-base already includes hi*8.
// halves 0..7  = W[k .. k+7]
// halves 8..15 = W[k+16 .. k+23]
__device__ __forceinline__ v16h load_a(const float* __restrict__ row, int k) {
  v16h a;
#pragma unroll
  for (int i = 0; i < 8; ++i) a[i] = (_Float16)row[k + i];
#pragma unroll
  for (int i = 0; i < 8; ++i) a[8 + i] = (_Float16)row[k + 16 + i];
  return a;
}

// B-matrix fragment (vector broadcast across all 16 columns): K values
// k..k+7 and k+16..k+23 (k already includes hi*8), matching load_a.
__device__ __forceinline__ v16h load_b(const float* __restrict__ v, int k) {
  v16h b;
#pragma unroll
  for (int i = 0; i < 8; ++i) b[i] = (_Float16)v[k + i];
#pragma unroll
  for (int i = 0; i < 8; ++i) b[8 + i] = (_Float16)v[k + 16 + i];
  return b;
}

// --- gfx1250 async memory <-> LDS ops (ASYNCcnt-tracked) --------------------

// LDS byte offset of a __shared__ object = low 32 bits of its generic address
// (addrspace(3) offset; flat LDS aperture keeps offset in addr[31:0]).
__device__ __forceinline__ unsigned lds_off(const void* p) {
  return (unsigned)(size_t)p;
}

__device__ __forceinline__ void async_load_b128(unsigned lds_dst,
                                                const void* gsrc) {
  asm volatile("global_load_async_to_lds_b128 %0, %1, off"
               :
               : "v"(lds_dst), "v"((unsigned long long)(size_t)gsrc)
               : "memory");
}

__device__ __forceinline__ void async_store_b128(void* gdst,
                                                 unsigned lds_src) {
  asm volatile("global_store_async_from_lds_b128 %0, %1, off"
               :
               : "v"((unsigned long long)(size_t)gdst), "v"(lds_src)
               : "memory");
}

__device__ __forceinline__ void wait_async0() {
  asm volatile("s_wait_asynccnt 0x0" ::: "memory");
}

// ---------------------------------------------------------------- threefry --

__device__ __forceinline__ unsigned rotl32(unsigned x, int r) {
  return (x << r) | (x >> (32 - r));
}

// Threefry-2x32, 20 rounds (JAX PRNG core).
__device__ __forceinline__ void threefry2x32(unsigned k0, unsigned k1,
                                             unsigned x0, unsigned x1,
                                             unsigned* o0, unsigned* o1) {
  const unsigned ks2 = k0 ^ k1 ^ 0x1BD11BDAu;
  const int ra[4] = {13, 15, 26, 6};
  const int rb[4] = {17, 29, 16, 24};
  x0 += k0; x1 += k1;
#pragma unroll
  for (int i = 0; i < 4; ++i) { x0 += x1; x1 = rotl32(x1, ra[i]); x1 ^= x0; }
  x0 += k1; x1 += ks2 + 1u;
#pragma unroll
  for (int i = 0; i < 4; ++i) { x0 += x1; x1 = rotl32(x1, rb[i]); x1 ^= x0; }
  x0 += ks2; x1 += k0 + 2u;
#pragma unroll
  for (int i = 0; i < 4; ++i) { x0 += x1; x1 = rotl32(x1, ra[i]); x1 ^= x0; }
  x0 += k0; x1 += k1 + 3u;
#pragma unroll
  for (int i = 0; i < 4; ++i) { x0 += x1; x1 = rotl32(x1, rb[i]); x1 ^= x0; }
  x0 += k1; x1 += ks2 + 4u;
#pragma unroll
  for (int i = 0; i < 4; ++i) { x0 += x1; x1 = rotl32(x1, ra[i]); x1 ^= x0; }
  x0 += ks2; x1 += k0 + 5u;
  *o0 = x0; *o1 = x1;
}

// bits -> uniform(tiny,1) -> Gumbel, matching jax.random.gumbel in f32.
__device__ __forceinline__ float gumbel_from(unsigned bits) {
  float f = __uint_as_float((bits >> 9) | 0x3f800000u) - 1.0f; // [0,1)
  const float tiny = 1.1754943508222875e-38f;
  float u = fmaxf(tiny, f * (1.0f - tiny) + tiny);
  return -logf(-logf(u));
}

__device__ __forceinline__ float sigmf(float x) { return 1.0f / (1.0f + expf(-x)); }

// ----------------------------------------------------------------- kernel ---

__global__ __launch_bounds__(1024, 1) void controller_kernel(
    const float* __restrict__ emb,     // [P, L+1, H]
    const float* __restrict__ w_ih,    // [P, 4H, H]
    const float* __restrict__ w_hh,    // [P, 4H, H]
    const float* __restrict__ b_ih,    // [P, 4H]
    const float* __restrict__ b_hh,    // [P, 4H]
    const float* __restrict__ w_attn1, // [P, H, H]
    const float* __restrict__ w_attn2, // [P, H, H]
    const float* __restrict__ w_index, // [P, H]
    const int*   __restrict__ seedp,   // [1]
    float* __restrict__ out,           // [258]
    float* __restrict__ anchors)       // ws: [NSLOT, H]
{
  __shared__ __align__(16) float s_x[HN];
  __shared__ __align__(16) float s_h[PN][HN];
  __shared__ __align__(16) float s_c[PN][HN];
  __shared__ __align__(16) float s_g[4 * HN];
  __shared__ __align__(16) float s_a2[HN];   // w_attn2 @ h
  __shared__ __align__(16) float s_cur[HN];  // w_attn1 @ h
  __shared__ float s_logits[NSLOT];
  __shared__ float s_vals[NSLOT];
  __shared__ int   s_node;
  __shared__ float s_ent, s_lp;

  const int tid  = (int)threadIdx.x;
  const int lane = tid & 31;
  const int wid  = tid >> 5;
  const int mrow = lane & 15;      // M row within tile for A / C
  const int hi   = lane >> 4;      // half-wave selector
  const int hi8  = hi * 8;         // A K-offset
  const int lo16 = hi * 8;         // B K-offset

  for (int i = tid; i < PN * HN; i += 1024) {
    (&s_h[0][0])[i] = 0.0f;
    (&s_c[0][0])[i] = 0.0f;
  }
  if (tid == 0) { s_node = 0; s_ent = 0.0f; s_lp = 0.0f; }
  __syncthreads();

  for (int step = -PN; step < LN * PN; ++step) {
    const bool warm = (step < 0);
    const int l = warm ? -1 : (step >> 2);
    const int p = warm ? (step + PN) : (step & 3);

    // --- phase A: stage x into LDS via async copy (ASYNCcnt path) ----------
    {
      int er, ec;
      if (warm || l == 0) { er = p; ec = LN; }
      else { int ni = s_node; er = ni & 3; ec = ni >> 2; }
      const float* src = emb + ((size_t)er * (LN + 1) + ec) * HN;
      if (tid < 256) {  // waves 0..7, EXEC all-ones, 256 lanes x 16B = 4KB
        async_load_b128(lds_off(s_x) + (unsigned)tid * 16u,
                        (const char*)src + (size_t)tid * 16u);
        wait_async0();
      }
    }
    __syncthreads();

    float* hv = s_h[p];
    float* cv = s_c[p];

    // --- phase B: gates = W_ih@x + W_hh@h + b_ih + b_hh (WMMA mat-vec) ------
    {
      const float* Wih = w_ih + (size_t)p * 4 * HN * HN;
      const float* Whh = w_hh + (size_t)p * 4 * HN * HN;
      const float* bi_ = b_ih + (size_t)p * 4 * HN;
      const float* bh_ = b_hh + (size_t)p * 4 * HN;
      for (int t = wid; t < (4 * HN) / 16; t += 32) {
        const float* r1 = Wih + (size_t)(t * 16 + mrow) * HN;
        const float* r2 = Whh + (size_t)(t * 16 + mrow) * HN;
        // prefetch next tile's rows (global_prefetch_b8)
        if (t + 32 < (4 * HN) / 16) {
          __builtin_prefetch(Wih + (size_t)((t + 32) * 16 + mrow) * HN, 0, 1);
          __builtin_prefetch(Whh + (size_t)((t + 32) * 16 + mrow) * HN, 0, 1);
        }
        v8f acc = {0.f, 0.f, 0.f, 0.f, 0.f, 0.f, 0.f, 0.f};
        for (int k = 0; k < HN; k += 32) {
          acc = wmma_f16(load_a(r1, k + hi8), load_b(s_x, k + lo16), acc);
          acc = wmma_f16(load_a(r2, k + hi8), load_b(hv,  k + lo16), acc);
        }
        if (mrow == 0) {                       // lanes 0 and 16 own rows
          int rb = t * 16 + hi * 8;
#pragma unroll
          for (int v = 0; v < 8; ++v)
            s_g[rb + v] = acc[v] + bi_[rb + v] + bh_[rb + v];
        }
      }
    }
    __syncthreads();

    // --- phase C: LSTM pointwise update (gate order i,f,g,o) ----------------
    {
      float gi = sigmf(s_g[tid]);
      float gf = sigmf(s_g[HN + tid]);
      float gg = tanhf(s_g[2 * HN + tid]);
      float go = sigmf(s_g[3 * HN + tid]);
      float c2 = gf * cv[tid] + gi * gg;
      cv[tid] = c2;
      hv[tid] = go * tanhf(c2);
    }
    __syncthreads();

    // --- phase D: attention mat-vecs (WMMA): cur = A1@h, a2 = A2@h ----------
    {
      const float* A1 = w_attn1 + (size_t)p * HN * HN;
      const float* A2 = w_attn2 + (size_t)p * HN * HN;
      for (int tt = wid; tt < 128; tt += 32) {
        const float* W = (tt < 64) ? A1 : A2;
        float* dst     = (tt < 64) ? s_cur : s_a2;
        int t = tt & 63;
        const float* r = W + (size_t)(t * 16 + mrow) * HN;
        v8f acc = {0.f, 0.f, 0.f, 0.f, 0.f, 0.f, 0.f, 0.f};
        for (int k = 0; k < HN; k += 32)
          acc = wmma_f16(load_a(r, k + hi8), load_b(hv, k + lo16), acc);
        if (mrow == 0) {
          int rb = t * 16 + hi * 8;
#pragma unroll
          for (int v = 0; v < 8; ++v) dst[rb + v] = acc[v];
        }
      }
    }
    __syncthreads();

    // --- phase E: append anchor (w_attn1@h) to global anchor list -----------
    // LDS -> global via async store (ASYNCcnt path).
    {
      int slot = warm ? p : (PN * (l + 1) + p);
      float* d = anchors + (size_t)slot * HN;
      if (tid < 256) {  // waves 0..7, 256 lanes x 16B = 4KB
        async_store_b128((char*)d + (size_t)tid * 16u,
                         lds_off(s_cur) + (unsigned)tid * 16u);
        wait_async0();
      }
    }
    __threadfence();   // make anchor rows visible across the WGP's L0s
    __syncthreads();

    if (!warm) {
      const int n = PN * (l + 1);

      // --- phase F: logits[j] = 2.5*tanh( dot(tanh(anchor_j + a2), w_idx)/5 )
      const float* wip = w_index + (size_t)p * HN;
      for (int j = wid; j < n; j += 32) {
        const float* aj = anchors + (size_t)j * HN;
        float part = 0.0f;
        for (int k = lane; k < HN; k += 32)
          part += tanhf(aj[k] + s_a2[k]) * wip[k];
#pragma unroll
        for (int off = 16; off > 0; off >>= 1)
          part += __shfl_xor(part, off, 32);
        if (lane == 0) s_logits[j] = 2.5f * tanhf(part * 0.2f);
      }
      __syncthreads();

      // --- phase G: categorical sample + entropy/logprob (thread 0) ---------
      if (tid == 0) {
        long long sd = (long long)seedp[0];
        unsigned bk0 = (unsigned)(((unsigned long long)sd) >> 32);
        unsigned bk1 = (unsigned)((unsigned long long)sd & 0xffffffffull);
        unsigned k0, k1;
        threefry2x32(bk0, bk1, 0u, (unsigned)step, &k0, &k1); // fold_in(key, l*P+p)

        int half = n >> 1;                   // n is a multiple of 4
        for (int i = 0; i < half; ++i) {
          unsigned o0, o1;
          threefry2x32(k0, k1, (unsigned)i, (unsigned)(half + i), &o0, &o1);
          s_vals[i]        = s_logits[i]        + gumbel_from(o0);
          s_vals[half + i] = s_logits[half + i] + gumbel_from(o1);
        }
        int   bidx = 0;
        float best = s_vals[0];
        for (int j = 1; j < n; ++j)
          if (s_vals[j] > best) { best = s_vals[j]; bidx = j; }

        float mx = s_logits[0];
        for (int j = 1; j < n; ++j) mx = fmaxf(mx, s_logits[j]);
        float se = 0.0f;
        for (int j = 0; j < n; ++j) se += expf(s_logits[j] - mx);
        float lse = mx + logf(se);
        float ent = 0.0f;
        for (int j = 0; j < n; ++j) {
          float lpj = s_logits[j] - lse;
          ent -= expf(lpj) * lpj;
        }
        s_ent += ent;
        s_lp  += -(s_logits[bidx] - lse);
        s_node = bidx;
        out[step]           = (float)(bidx & 3);   // arch % P
        out[LN * PN + step] = (float)(bidx >> 2);  // arch // P
      }
    }
    __syncthreads();
  }

  if (tid == 0) {
    out[2 * LN * PN]     = s_ent;
    out[2 * LN * PN + 1] = s_lp;
  }
}

// ----------------------------------------------------------------- launch ---

extern "C" void kernel_launch(void* const* d_in, const int* in_sizes, int n_in,
                              void* d_out, int out_size, void* d_ws, size_t ws_size,
                              hipStream_t stream) {
  (void)in_sizes; (void)n_in; (void)out_size; (void)ws_size;
  const float* emb    = (const float*)d_in[0];
  const float* w_ih   = (const float*)d_in[1];
  const float* w_hh   = (const float*)d_in[2];
  const float* b_ih   = (const float*)d_in[3];
  const float* b_hh   = (const float*)d_in[4];
  const float* w_a1   = (const float*)d_in[5];
  const float* w_a2   = (const float*)d_in[6];
  const float* w_idx  = (const float*)d_in[7];
  const int*   seed   = (const int*)d_in[8];
  float* out     = (float*)d_out;
  float* anchors = (float*)d_ws;   // NSLOT * HN floats (~540 KB)

  controller_kernel<<<1, 1024, 0, stream>>>(emb, w_ih, w_hh, b_ih, b_hh,
                                            w_a1, w_a2, w_idx, seed,
                                            out, anchors);
}
